// RnnExtractor_8581344658204
// MI455X (gfx1250) — matrix operand order repaired and checked by reading, not compiled, not run on hardware
//
#include <hip/hip_runtime.h>
#include <hip/hip_bf16.h>
#include <math.h>

// ---------------------------------------------------------------------------
// LayerNorm-LSTM 3-cell scan for MI455X (gfx1250), wave32 + WMMA f16->f32.
//
// Mapping: 12 persistent workgroups = 3 cells x 4 row-blocks (16 batch rows).
// Each WG: 512 threads = 16 wave32s; per step each wave computes 8 N-tiles of
// the two GEMM panels ([16,K]x[K,2048]) with v_wmma_f32_16x16x32_f16, storing
// f16 pre-LN products to LDS; LN stats / gates / c,h updates are f32 in LDS.
// One grid barrier per timestep (shared publishes h_s(t) into a double buffer,
// pi/vf consume it while shared runs t+1). h_s staging uses CDNA5 async
// global->LDS copies (ASYNCcnt) overlapped with the done-mask pass.
// ---------------------------------------------------------------------------

typedef __attribute__((ext_vector_type(16))) _Float16 v16h;
typedef __attribute__((ext_vector_type(8)))  _Float16 v8h;
typedef __attribute__((ext_vector_type(8)))  float    v8f;

#define T_STEPS 1024
#define BATCH   64
#define FEAT    256
#define HID     512
#define NG      2048      // 4*HID gate width
#define NWG     12
#define TPB     512
#define ROWS    16        // batch rows per WG == one WMMA M tile
#define NTILES  128       // NG / 16

struct LnParams { const float *b, *gx, *bx, *gh, *bh, *gc, *bc; };

// ---------------- weight convert: f32 [K][2048] -> f16 [2048][K] -----------
__global__ void wconvert_kernel(const float* __restrict__ src,
                                _Float16* __restrict__ dst, int K) {
  int idx = blockIdx.x * blockDim.x + threadIdx.x;
  int total = NG * K;
  if (idx < total) {
    int n = idx / K, k = idx - n * K;
    dst[idx] = (_Float16)src[(size_t)k * NG + n];
  }
}

// ---------------- helpers ---------------------------------------------------
__device__ __forceinline__ float sigm(float x) { return 1.f / (1.f + __expf(-x)); }

__device__ __forceinline__ void grid_sync(unsigned* cnt, unsigned target) {
  __threadfence();                       // release stores (h_s publish)
  __syncthreads();
  if (threadIdx.x == 0) {
    atomicAdd(cnt, 1u);
    while (atomicAdd(cnt, 0u) < target) __builtin_amdgcn_s_sleep(1);
  }
  __syncthreads();
  __threadfence();                       // acquire for subsequent loads
}

// A fragment, 16x32, this code's contiguous k order (not the ISA table order): lanes 0-15 rows M, K = {kb..kb+15},
// kb = 0 (lanes<16) / 16 (lanes>=16). Two 16B LDS loads.
__device__ __forceinline__ v16h frag_a(const _Float16* A, int ldk, int kt, int lane) {
  int m  = lane & 15;
  int kb = (lane & 16) ? 16 : 0;
  const _Float16* p = A + m * ldk + kt * 32 + kb;
  v8h lo = *(const v8h*)(p);
  v8h hi = *(const v8h*)(p + 8);
  return __builtin_shufflevector(lo, hi, 0,1,2,3,4,5,6,7,8,9,10,11,12,13,14,15);
}

// B fragment, 32x16, same contiguous k order as frag_a: lane holds column n = nt*16 + (lane&15) and 16
// consecutive K values (0..15 for lanes<16, 16..31 otherwise). With weights
// stored [N][K] this is one contiguous 32B load straight from L2.
__device__ __forceinline__ v16h frag_b(const _Float16* __restrict__ Wt, int K,
                                       int nt, int kt, int lane) {
  int n  = nt * 16 + (lane & 15);
  int ks = kt * 32 + ((lane & 16) ? 16 : 0);
  return *(const v16h*)(Wt + (size_t)n * K + ks);
}

// One GEMM panel: [16,K] (LDS f16) x [K,2048] (global f16, [N][K]) -> P (LDS f16)
__device__ __forceinline__ void gemm_panel(const _Float16* A, int K,
                                           const _Float16* __restrict__ Wt,
                                           _Float16* Pl, int wv, int lane) {
  // wv is wave-uniform (readfirstlane'd by caller) -> scalar loop, EXEC stays
  // all-ones around WMMA as the ISA requires.
  for (int nt = wv; nt < NTILES; nt += 16) {
    if (nt + 16 < NTILES)
      __builtin_prefetch(Wt + (size_t)(nt + 16) * 16 * K, 0, 1);  // global_prefetch_b8
    v8f acc = {};
    const int kts = K >> 5;                    // K/32 (8 or 16, always even)
    for (int kt = 0; kt < kts; kt += 2) {      // unroll x2 for WMMA issue pipelining
      v16h a0 = frag_a(A, K, kt, lane);
      v16h b0 = frag_b(Wt, K, nt, kt, lane);
      acc = __builtin_amdgcn_wmma_f32_16x16x32_f16(false, a0, false, b0,
                                                   (short)0, acc, false, false);
      v16h a1 = frag_a(A, K, kt + 1, lane);
      v16h b1 = frag_b(Wt, K, nt, kt + 1, lane);
      acc = __builtin_amdgcn_wmma_f32_16x16x32_f16(false, a1, false, b1,
                                                   (short)0, acc, false, false);
    }
    // C/D layout: vgpr v, lanes<16 -> row v, lanes>=16 -> row v+8; col = lane&15
    int nn = nt * 16 + (lane & 15);
    int rB = (lane & 16) ? 8 : 0;
#pragma unroll
    for (int v = 0; v < 8; ++v)
      Pl[(size_t)(rB + v) * NG + nn] = (_Float16)acc[v];
  }
}

__device__ __forceinline__ void row_stats_f16(const _Float16* Pl, int ncols,
                                              float* sm, float* sr, int wv, int lane) {
  float s = 0.f, ss = 0.f;
  const _Float16* row = Pl + (size_t)wv * ncols;
  for (int j = lane; j < ncols; j += 32) { float v = (float)row[j]; s += v; ss += v * v; }
#pragma unroll
  for (int o = 16; o > 0; o >>= 1) { s += __shfl_xor(s, o, 32); ss += __shfl_xor(ss, o, 32); }
  if (lane == 0) {
    float inv = 1.f / (float)ncols;
    float m = s * inv;
    float var = fmaxf(ss * inv - m * m, 0.f);
    sm[wv] = m; sr[wv] = rsqrtf(var + 1e-5f);
  }
}

__device__ __forceinline__ void row_stats_f32(const float* Cl, int ncols,
                                              float* sm, float* sr, int wv, int lane) {
  float s = 0.f, ss = 0.f;
  const float* row = Cl + (size_t)wv * ncols;
  for (int j = lane; j < ncols; j += 32) { float v = row[j]; s += v; ss += v * v; }
#pragma unroll
  for (int o = 16; o > 0; o >>= 1) { s += __shfl_xor(s, o, 32); ss += __shfl_xor(ss, o, 32); }
  if (lane == 0) {
    float inv = 1.f / (float)ncols;
    float m = s * inv;
    float var = fmaxf(ss * inv - m * m, 0.f);
    sm[wv] = m; sr[wv] = rsqrtf(var + 1e-5f);
  }
}

// ---------------- persistent scan kernel ------------------------------------
__global__ void __launch_bounds__(TPB)
lnlstm_scan_kernel(const float* __restrict__ feat,
                   const float* __restrict__ dones,
                   const _Float16* __restrict__ Wxs, const _Float16* __restrict__ Whs,
                   const _Float16* __restrict__ Wxp, const _Float16* __restrict__ Whp,
                   const _Float16* __restrict__ Wxv, const _Float16* __restrict__ Whv,
                   LnParams ps, LnParams pp, LnParams pv,
                   _Float16* __restrict__ hsbuf,       // double-buffered h_s, f16
                   unsigned* __restrict__ sync_cnt,
                   float* __restrict__ out) {
  extern __shared__ char smem[];
  _Float16* P1  = (_Float16*)smem;          // [16][2048] pre-LN x@Wx
  _Float16* P2  = P1 + ROWS * NG;           // [16][2048] pre-LN h@Wh
  _Float16* A1  = P2 + ROWS * NG;           // [16][<=512] staged GEMM input x
  _Float16* A2  = A1 + ROWS * HID;          // [16][512]  recurrent h (f16)
  float*    Cs  = (float*)(A2 + ROWS * HID);// [16][512]  cell state c (f32)
  float*    sm1 = Cs + ROWS * HID;          // LN stats (6 x 16)
  float*    sr1 = sm1 + ROWS;
  float*    sm2 = sr1 + ROWS;
  float*    sr2 = sm2 + ROWS;
  float*    smc = sr2 + ROWS;
  float*    srv = smc + ROWS;

  const int wg   = blockIdx.x;
  const int cell = wg >> 2;            // 0=shared, 1=pi, 2=vf
  const int rb   = wg & 3;             // 16-row batch block
  const int tid  = threadIdx.x;
  const int wv   = __builtin_amdgcn_readfirstlane(tid >> 5); // uniform wave id
  const int lane = tid & 31;
  const int col  = tid;                // hidden column 0..511

  const _Float16* W1 = (cell == 0) ? Wxs : (cell == 1) ? Wxp : Wxv;
  const _Float16* W2 = (cell == 0) ? Whs : (cell == 1) ? Whp : Whv;
  const int K1 = (cell == 0) ? FEAT : HID;
  const LnParams P = (cell == 0) ? ps : (cell == 1) ? pp : pv;
  float* outb = (cell == 2) ? (out + (size_t)T_STEPS * BATCH * HID) : out;

  // wave-relative LDS byte offset of A1 (hardware adds LDS_BASE for async ops)
  const unsigned a1_lds = (unsigned)(uintptr_t)A1;

  // Column-constant LN/bias params live in registers for the whole scan.
  float pb[4], pgx[4], pbx[4], pgh[4], pbh[4];
#pragma unroll
  for (int g = 0; g < 4; ++g) {
    int cc = col + g * HID;
    pb[g] = P.b[cc]; pgx[g] = P.gx[cc]; pbx[g] = P.bx[cc];
    pgh[g] = P.gh[cc]; pbh[g] = P.bh[cc];
  }
  const float gcv = P.gc[col], bcv = P.bc[col];

  for (int i = tid; i < ROWS * HID; i += TPB) { A2[i] = (_Float16)0.f; Cs[i] = 0.f; }
  __syncthreads();

  for (int t = 0; t < T_STEPS; ++t) {
    if (cell != 0) {
      // rendezvous: shared has published h_s(t) into hsbuf[t&1]; then kick off
      // async global->LDS staging of the [16,512] f16 h_s tile (ASYNCcnt),
      // overlapped with the done-mask pass below (touches only A2/Cs).
      grid_sync(sync_cnt, (unsigned)(t + 1) * NWG);
      const char* hsrc = (const char*)(hsbuf + ((size_t)(t & 1) * BATCH + rb * ROWS) * HID);
      unsigned o = (unsigned)tid * 32u;                 // 512 thr x 32B = 16KB
      unsigned la = a1_lds + o;
      unsigned long long ga = (unsigned long long)(uintptr_t)(hsrc + o);
      asm volatile("global_load_async_to_lds_b128 %0, %1, off"
                   :: "v"(la), "v"(ga) : "memory");
      asm volatile("global_load_async_to_lds_b128 %0, %1, off offset:16"
                   :: "v"(la), "v"(ga) : "memory");
    }

    // ---- reset hiddens of done envs (mask = 1 - dones[t,b]) ----
#pragma unroll 4
    for (int i = 0; i < ROWS; ++i) {
      float m = 1.f - dones[(size_t)t * BATCH + rb * ROWS + i];
      int idx = i * HID + col;
      Cs[idx] *= m;
      A2[idx] = (_Float16)((float)A2[idx] * m);
    }

    if (cell == 0) {
      // stage x(t) tile (16 x 256) as f16 (f32 source needs conversion)
      const float* xs = feat + ((size_t)t * BATCH + rb * ROWS) * FEAT;
      for (int i = tid; i < ROWS * FEAT; i += TPB) A1[i] = (_Float16)xs[i];
    } else {
      asm volatile("s_wait_asynccnt 0x0" ::: "memory");  // h_s tile landed in LDS
    }
    __syncthreads();

    // ---- WMMA GEMM panels (f32 accumulate) ----
    gemm_panel(A1, K1,  W1, P1, wv, lane);
    gemm_panel(A2, HID, W2, P2, wv, lane);
    __syncthreads();

    // ---- per-row LayerNorm stats over the 2048 gate columns ----
    row_stats_f16(P1, NG, sm1, sr1, wv, lane);
    row_stats_f16(P2, NG, sm2, sr2, wv, lane);
    __syncthreads();

    // ---- gates i,f,g -> c update ----
    for (int i = 0; i < ROWS; ++i) {
      float m1 = sm1[i], r1 = sr1[i], m2 = sm2[i], r2 = sr2[i];
      float pre[3];
#pragma unroll
      for (int g = 0; g < 3; ++g) {
        int cc = col + g * HID;
        float p1 = (float)P1[i * NG + cc];
        float p2 = (float)P2[i * NG + cc];
        pre[g] = (p1 - m1) * r1 * pgx[g] + pbx[g]
               + (p2 - m2) * r2 * pgh[g] + pbh[g] + pb[g];
      }
      float cn = sigm(pre[1]) * Cs[i * HID + col] + sigm(pre[0]) * tanhf(pre[2]);
      Cs[i * HID + col] = cn;
    }
    __syncthreads();

    // ---- LN stats of c over 512 ----
    row_stats_f32(Cs, HID, smc, srv, wv, lane);
    __syncthreads();

    // ---- output gate -> h; publish ----
    for (int i = 0; i < ROWS; ++i) {
      float m1 = sm1[i], r1 = sr1[i], m2 = sm2[i], r2 = sr2[i];
      int cc = col + 3 * HID;
      float p1 = (float)P1[i * NG + cc];
      float p2 = (float)P2[i * NG + cc];
      float og = (p1 - m1) * r1 * pgx[3] + pbx[3]
               + (p2 - m2) * r2 * pgh[3] + pbh[3] + pb[3];
      float cn = Cs[i * HID + col];
      float hn = sigm(og) * tanhf((cn - smc[i]) * srv[i] * gcv + bcv);
      A2[i * HID + col] = (_Float16)hn;
      if (cell == 0)
        hsbuf[((size_t)(t & 1) * BATCH + rb * ROWS + i) * HID + col] = (_Float16)hn;
      else
        outb[((size_t)t * BATCH + rb * ROWS + i) * HID + col] = hn;
    }

    if (cell == 0) {
      // release h_s(t); pi/vf of step t run while shared starts t+1.
      grid_sync(sync_cnt, (unsigned)(t + 1) * NWG);
    }
    __syncthreads();
  }
}

// ---------------- launch -----------------------------------------------------
extern "C" void kernel_launch(void* const* d_in, const int* in_sizes, int n_in,
                              void* d_out, int out_size, void* d_ws, size_t ws_size,
                              hipStream_t stream) {
  (void)in_sizes; (void)n_in; (void)out_size; (void)ws_size;
  const float* feat  = (const float*)d_in[0];
  const float* dones = (const float*)d_in[1];

  // param dicts flatten as: Wx, Wh, b, gx, bx, gh, bh, gc, bc
  auto prm = [&](int base) {
    return LnParams{ (const float*)d_in[base + 2], (const float*)d_in[base + 3],
                     (const float*)d_in[base + 4], (const float*)d_in[base + 5],
                     (const float*)d_in[base + 6], (const float*)d_in[base + 7],
                     (const float*)d_in[base + 8] };
  };

  // workspace layout (f16 elements): 6 transposed weight matrices + h_s x2 + counter
  _Float16* ws16  = (_Float16*)d_ws;
  _Float16* Wxs   = ws16 + 0;        // 2048*256
  _Float16* Whs   = ws16 + 524288;   // 2048*512
  _Float16* Wxp   = ws16 + 1572864;
  _Float16* Whp   = ws16 + 2621440;
  _Float16* Wxv   = ws16 + 3670016;
  _Float16* Whv   = ws16 + 4718592;
  _Float16* hsbuf = ws16 + 5767168;  // 2 * 64 * 512
  unsigned* sync_cnt = (unsigned*)((char*)d_ws + 11665408);

  hipMemsetAsync(sync_cnt, 0, 256, stream);

  const int thr = 256;
  wconvert_kernel<<<(NG * FEAT + thr - 1) / thr, thr, 0, stream>>>((const float*)d_in[2],  Wxs, FEAT);
  wconvert_kernel<<<(NG * HID  + thr - 1) / thr, thr, 0, stream>>>((const float*)d_in[3],  Whs, HID);
  wconvert_kernel<<<(NG * HID  + thr - 1) / thr, thr, 0, stream>>>((const float*)d_in[11], Wxp, HID);
  wconvert_kernel<<<(NG * HID  + thr - 1) / thr, thr, 0, stream>>>((const float*)d_in[12], Whp, HID);
  wconvert_kernel<<<(NG * HID  + thr - 1) / thr, thr, 0, stream>>>((const float*)d_in[20], Wxv, HID);
  wconvert_kernel<<<(NG * HID  + thr - 1) / thr, thr, 0, stream>>>((const float*)d_in[21], Whv, HID);

  // dynamic LDS: P1,P2 (128KB f16) + A1,A2 (32KB f16) + Cs (32KB f32) + stats
  size_t smem_bytes = (size_t)ROWS * NG * 2 * sizeof(_Float16)
                    + (size_t)ROWS * HID * 2 * sizeof(_Float16)
                    + (size_t)ROWS * HID * sizeof(float)
                    + 6 * ROWS * sizeof(float);

  lnlstm_scan_kernel<<<NWG, TPB, smem_bytes, stream>>>(
      feat, dones, Wxs, Whs, Wxp, Whp, Wxv, Whv,
      prm(2), prm(11), prm(20), hsbuf, sync_cnt, (float*)d_out);
}
